// SAttention_7215545057874
// MI455X (gfx1250) — compile-verified, matrix operand order checked by reading, not run on hardware
//
#include <hip/hip_runtime.h>

#define SP    4128      // padded sequence (multiple of 32), real S = 4097
#define SEQ   4097
#define CDIM  512
#define HEADS 8
#define DHEAD 64
#define KTAPS 33

typedef _Float16 v16h __attribute__((ext_vector_type(16)));
typedef _Float16 v8h  __attribute__((ext_vector_type(8)));
typedef float    v8f  __attribute__((ext_vector_type(8)));

union HFrag { v16h v; v8h h[2]; };

// Load a 16x32 f16 WMMA operand fragment.
// A-operand: idx0 = row base (rows contiguous in K).
// B-operand: idx0 = column base, where each column is stored K-contiguous
// (weights row-major over K; K matrix token-major over d; V stored transposed).
__device__ __forceinline__ v16h load_frag(const _Float16* __restrict__ base,
                                          int ld, int idx0, int k0) {
  int lane = threadIdx.x & 31;
  int idx  = idx0 + (lane & 15);
  int hf   = (lane >> 4) * 8;
  const _Float16* p = base + (size_t)idx * ld + k0 + hf;
  HFrag u;
  u.h[0] = *(const v8h*)(p);
  u.h[1] = *(const v8h*)(p + 16);
  return u.v;
}

__device__ __forceinline__ v8f wmma16(v16h a, v16h b, v8f c) {
  return __builtin_amdgcn_wmma_f32_16x16x32_f16(false, a, false, b,
                                                (short)0, c, false, false);
}

// ---------------- input assembly ----------------
__global__ __launch_bounds__(256) void build_h(const float* __restrict__ x,
                                               const float* __restrict__ cls,
                                               float* __restrict__ h) {
  int i = blockIdx.x * 256 + threadIdx.x;
  if (i >= SP * CDIM) return;
  int row = i >> 9, c = i & (CDIM - 1);
  float v;
  if (row == 0)        v = cls[c];
  else if (row < SEQ)  v = x[(size_t)(row - 1) * CDIM + c];
  else                 v = 0.f;
  h[i] = v;
}

__global__ __launch_bounds__(256) void f32_to_f16(const float* __restrict__ s,
                                                  _Float16* __restrict__ d, int n) {
  int i = blockIdx.x * 256 + threadIdx.x;
  if (i < n) d[i] = (_Float16)s[i];
}

// ---------------- per-row LayerNorm -> f16 ----------------
__global__ __launch_bounds__(256) void ln_rows(const float* __restrict__ h,
                                               const float* __restrict__ g,
                                               _Float16* __restrict__ hln) {
  __shared__ float red[256];
  int row = blockIdx.x, t = threadIdx.x;
  const float* xr = h + (size_t)row * CDIM;
  float a = 0.f, b = 0.f;
  if (row < SEQ) { a = xr[t]; b = xr[t + 256]; }
  red[t] = a + b; __syncthreads();
  #pragma unroll
  for (int off = 128; off > 0; off >>= 1) {
    if (t < off) red[t] += red[t + off];
    __syncthreads();
  }
  float mean = red[0] * (1.f / CDIM);
  __syncthreads();
  float da = a - mean, db = b - mean;
  red[t] = da * da + db * db; __syncthreads();
  #pragma unroll
  for (int off = 128; off > 0; off >>= 1) {
    if (t < off) red[t] += red[t + off];
    __syncthreads();
  }
  float rs = rsqrtf(red[0] * (1.f / CDIM) + 1e-5f);
  if (row < SEQ) {
    hln[(size_t)row * CDIM + t]       = (_Float16)(da * rs * g[t]);
    hln[(size_t)row * CDIM + t + 256] = (_Float16)(db * rs * g[t + 256]);
  } else {
    hln[(size_t)row * CDIM + t]       = (_Float16)0.f;
    hln[(size_t)row * CDIM + t + 256] = (_Float16)0.f;
  }
}

// ---------------- QKV GEMM: [SP,512] x [512,1536]^T ----------------
// One wave -> 32x32 output tile (2x2 WMMA accumulators), K=512 in 16 steps.
// Epilogue scatters into Q (token-major, pre-scaled), K (token-major),
// V^T (d-major, for contiguous PV B-operand + conv).
__global__ __launch_bounds__(256) void gemm_qkv(const _Float16* __restrict__ hln,
                                                const _Float16* __restrict__ w16,
                                                _Float16* __restrict__ qh,
                                                _Float16* __restrict__ kh,
                                                _Float16* __restrict__ vT) {
  const int NT = (3 * CDIM) / 32;                       // 48 column tiles
  int wave = (blockIdx.x * blockDim.x + threadIdx.x) >> 5;
  int lane = threadIdx.x & 31;
  int m0 = (wave / NT) * 32, n0 = (wave % NT) * 32;
  v8f acc[2][2] = {};
  for (int k0 = 0; k0 < CDIM; k0 += 32) {
    if (k0 + 32 < CDIM)
      __builtin_prefetch(hln + (size_t)(m0 + lane) * CDIM + k0 + 32, 0, 1);
    v16h a0 = load_frag(hln, CDIM, m0,      k0);
    v16h a1 = load_frag(hln, CDIM, m0 + 16, k0);
    v16h b0 = load_frag(w16, CDIM, n0,      k0);
    v16h b1 = load_frag(w16, CDIM, n0 + 16, k0);
    acc[0][0] = wmma16(a0, b0, acc[0][0]);
    acc[0][1] = wmma16(a0, b1, acc[0][1]);
    acc[1][0] = wmma16(a1, b0, acc[1][0]);
    acc[1][1] = wmma16(a1, b1, acc[1][1]);
  }
  int cl = lane & 15, hf = lane >> 4;
  #pragma unroll
  for (int im = 0; im < 2; ++im)
    #pragma unroll
    for (int in = 0; in < 2; ++in)
      #pragma unroll
      for (int r = 0; r < 8; ++r) {
        int row = m0 + im * 16 + r + 8 * hf;
        int col = n0 + in * 16 + cl;
        float val = acc[im][in][r];
        int which = col >> 9;
        int head  = (col >> 6) & 7;
        int d     = col & 63;
        if (which == 0)
          qh[((size_t)head * SP + row) * DHEAD + d] = (_Float16)(val * 0.125f);
        else if (which == 1)
          kh[((size_t)head * SP + row) * DHEAD + d] = (_Float16)val;
        else
          vT[(size_t)(head * DHEAD + d) * SP + row] = (_Float16)val;
      }
}

// ---------------- flash attention: one wave = 16 queries of one head -------
__global__ __launch_bounds__(256) void attn_fa(const _Float16* __restrict__ qh,
                                               const _Float16* __restrict__ kh,
                                               const _Float16* __restrict__ vT,
                                               _Float16* __restrict__ o16) {
  __shared__ __align__(16) _Float16 plds[8][16 * 32];   // P staging per wave
  int wl   = threadIdx.x >> 5;
  int lane = threadIdx.x & 31;
  int wave = blockIdx.x * 8 + wl;
  const int QT = SP / 16;                               // 258
  int head = wave / QT;
  int q0   = (wave % QT) * 16;
  const _Float16* Q = qh + (size_t)head * SP * DHEAD;
  const _Float16* K = kh + (size_t)head * SP * DHEAD;
  const _Float16* V = vT + (size_t)head * DHEAD * SP;
  _Float16* myp = plds[wl];
  int cl = lane & 15, hf = lane >> 4;

  v16h aq0 = load_frag(Q, DHEAD, q0, 0);
  v16h aq1 = load_frag(Q, DHEAD, q0, 32);
  v8f o_acc[4] = {};
  float m_r[8], l_r[8];
  #pragma unroll
  for (int r = 0; r < 8; ++r) { m_r[r] = -1e30f; l_r[r] = 0.f; }

  for (int kb = 0; kb < SP; kb += 32) {
    v8f s0 = {}, s1 = {};
    s0 = wmma16(aq0, load_frag(K, DHEAD, kb,      0),  s0);
    s0 = wmma16(aq1, load_frag(K, DHEAD, kb,      32), s0);
    s1 = wmma16(aq0, load_frag(K, DHEAD, kb + 16, 0),  s1);
    s1 = wmma16(aq1, load_frag(K, DHEAD, kb + 16, 32), s1);
    bool ok0 = (kb + cl)      < SEQ;
    bool ok1 = (kb + 16 + cl) < SEQ;
    float alpha[8];
    #pragma unroll
    for (int r = 0; r < 8; ++r) {
      float e0 = ok0 ? s0[r] : -1e30f;
      float e1 = ok1 ? s1[r] : -1e30f;
      float mx = fmaxf(e0, e1);
      #pragma unroll
      for (int off = 8; off > 0; off >>= 1)
        mx = fmaxf(mx, __shfl_xor(mx, off, 16));
      float mnew = fmaxf(m_r[r], mx);
      float p0 = __expf(e0 - mnew);
      float p1 = __expf(e1 - mnew);
      float a  = __expf(m_r[r] - mnew);
      float sm = p0 + p1;
      #pragma unroll
      for (int off = 8; off > 0; off >>= 1)
        sm += __shfl_xor(sm, off, 16);
      l_r[r] = l_r[r] * a + sm;
      m_r[r] = mnew;
      alpha[r] = a;
      int row = r + 8 * hf;
      myp[row * 32 + cl]      = (_Float16)p0;       // C-layout -> LDS
      myp[row * 32 + 16 + cl] = (_Float16)p1;
    }
    #pragma unroll
    for (int t = 0; t < 4; ++t)
      #pragma unroll
      for (int r = 0; r < 8; ++r) o_acc[t][r] *= alpha[r];
    asm volatile("s_wait_dscnt 0" ::: "memory");      // P visible before A-frag read
    HFrag up;                                         // LDS -> A-layout fragment
    const _Float16* prow = myp + cl * 32;
    up.h[0] = *(const v8h*)(prow + hf * 8);
    up.h[1] = *(const v8h*)(prow + 16 + hf * 8);
    v16h pfrag = up.v;
    #pragma unroll
    for (int t = 0; t < 4; ++t)
      o_acc[t] = wmma16(pfrag, load_frag(V, SP, t * 16, kb), o_acc[t]);
  }
  #pragma unroll
  for (int t = 0; t < 4; ++t)
    #pragma unroll
    for (int r = 0; r < 8; ++r) {
      int row = q0 + r + 8 * hf;
      int col = head * DHEAD + t * 16 + cl;
      o16[(size_t)row * CDIM + col] = (_Float16)(o_acc[t][r] / l_r[r]);
    }
}

// ---------------- depthwise conv residual on V (33 taps over sequence) -----
__global__ __launch_bounds__(256) void conv_res(const _Float16* __restrict__ vT,
                                                const float* __restrict__ resk,
                                                _Float16* __restrict__ o16) {
  const int TOT = HEADS * DHEAD * SP;
  int idx = blockIdx.x * 256 + threadIdx.x;
  if (idx >= TOT) return;
  int n  = idx % SP;
  int hd = idx / SP;                    // head*64 + d
  int head = hd >> 6;
  const _Float16* src = vT + (size_t)hd * SP;
  float acc = 0.f;
  #pragma unroll
  for (int t = 0; t < KTAPS; ++t) {
    int n2 = n + t - KTAPS / 2;
    if (n2 >= 0 && n2 < SP) acc += (float)src[n2] * resk[head * KTAPS + t];
  }
  size_t oi = (size_t)n * CDIM + hd;
  o16[oi] = (_Float16)((float)o16[oi] + acc);
}

// ---------------- projection GEMM + residual: h += O * Wp^T + b ------------
__global__ __launch_bounds__(256) void gemm_proj(const _Float16* __restrict__ o16,
                                                 const _Float16* __restrict__ w16,
                                                 const float* __restrict__ bproj,
                                                 float* __restrict__ h) {
  const int NT = CDIM / 32;                             // 16 column tiles
  int wave = (blockIdx.x * blockDim.x + threadIdx.x) >> 5;
  int lane = threadIdx.x & 31;
  int m0 = (wave / NT) * 32, n0 = (wave % NT) * 32;
  v8f acc[2][2] = {};
  for (int k0 = 0; k0 < CDIM; k0 += 32) {
    v16h a0 = load_frag(o16, CDIM, m0,      k0);
    v16h a1 = load_frag(o16, CDIM, m0 + 16, k0);
    v16h b0 = load_frag(w16, CDIM, n0,      k0);
    v16h b1 = load_frag(w16, CDIM, n0 + 16, k0);
    acc[0][0] = wmma16(a0, b0, acc[0][0]);
    acc[0][1] = wmma16(a0, b1, acc[0][1]);
    acc[1][0] = wmma16(a1, b0, acc[1][0]);
    acc[1][1] = wmma16(a1, b1, acc[1][1]);
  }
  int cl = lane & 15, hf = lane >> 4;
  #pragma unroll
  for (int im = 0; im < 2; ++im)
    #pragma unroll
    for (int in = 0; in < 2; ++in)
      #pragma unroll
      for (int r = 0; r < 8; ++r) {
        int row = m0 + im * 16 + r + 8 * hf;
        int col = n0 + in * 16 + cl;
        size_t hi = (size_t)row * CDIM + col;
        h[hi] = h[hi] + acc[im][in][r] + bproj[col];
      }
}

// ---------------- final LN of CLS row -> f32 output ------------------------
__global__ __launch_bounds__(256) void final_ln(const float* __restrict__ h,
                                                const float* __restrict__ g,
                                                float* __restrict__ out) {
  __shared__ float red[256];
  int t = threadIdx.x;
  float a = h[t], b = h[t + 256];
  red[t] = a + b; __syncthreads();
  #pragma unroll
  for (int off = 128; off > 0; off >>= 1) {
    if (t < off) red[t] += red[t + off];
    __syncthreads();
  }
  float mean = red[0] * (1.f / CDIM);
  __syncthreads();
  float da = a - mean, db = b - mean;
  red[t] = da * da + db * db; __syncthreads();
  #pragma unroll
  for (int off = 128; off > 0; off >>= 1) {
    if (t < off) red[t] += red[t + off];
    __syncthreads();
  }
  float rs = rsqrtf(red[0] * (1.f / CDIM) + 1e-5f);
  out[t]       = da * rs * g[t];
  out[t + 256] = db * rs * g[t + 256];
}

extern "C" void kernel_launch(void* const* d_in, const int* in_sizes, int n_in,
                              void* d_out, int out_size, void* d_ws, size_t ws_size,
                              hipStream_t stream) {
  const float* x     = (const float*)d_in[0];
  const float* cls   = (const float*)d_in[1];
  const float* ln_g[2]  = {(const float*)d_in[2],  (const float*)d_in[7]};
  const float* wqkv[2]  = {(const float*)d_in[3],  (const float*)d_in[8]};
  const float* wproj[2] = {(const float*)d_in[4],  (const float*)d_in[9]};
  const float* bproj[2] = {(const float*)d_in[5],  (const float*)d_in[10]};
  const float* resk[2]  = {(const float*)d_in[6],  (const float*)d_in[11]};
  const float* fc_g  = (const float*)d_in[12];
  float* out = (float*)d_out;

  char* p = (char*)d_ws;
  float*    h    = (float*)p;    p += (size_t)SP * CDIM * 4;
  _Float16* hln  = (_Float16*)p; p += (size_t)SP * CDIM * 2;
  _Float16* qh   = (_Float16*)p; p += (size_t)SP * CDIM * 2;   // H*SP*D == SP*C
  _Float16* kh   = (_Float16*)p; p += (size_t)SP * CDIM * 2;
  _Float16* vT   = (_Float16*)p; p += (size_t)SP * CDIM * 2;
  _Float16* o16  = (_Float16*)p; p += (size_t)SP * CDIM * 2;
  _Float16* wq16 = (_Float16*)p; p += (size_t)3 * CDIM * CDIM * 2;
  _Float16* wp16 = (_Float16*)p; p += (size_t)CDIM * CDIM * 2;

  build_h<<<(SP * CDIM + 255) / 256, 256, 0, stream>>>(x, cls, h);

  for (int L = 0; L < 2; ++L) {
    f32_to_f16<<<(3 * CDIM * CDIM + 255) / 256, 256, 0, stream>>>(wqkv[L], wq16,
                                                                  3 * CDIM * CDIM);
    f32_to_f16<<<(CDIM * CDIM + 255) / 256, 256, 0, stream>>>(wproj[L], wp16,
                                                              CDIM * CDIM);
    ln_rows<<<SP, 256, 0, stream>>>(h, ln_g[L], hln);
    gemm_qkv<<<(SP / 32) * (3 * CDIM / 32) / 8, 256, 0, stream>>>(hln, wq16,
                                                                  qh, kh, vT);
    attn_fa<<<HEADS * (SP / 16) / 8, 256, 0, stream>>>(qh, kh, vT, o16);
    conv_res<<<(HEADS * DHEAD * SP + 255) / 256, 256, 0, stream>>>(vT, resk[L], o16);
    gemm_proj<<<(SP / 32) * (CDIM / 32) / 8, 256, 0, stream>>>(o16, wp16,
                                                               bproj[L], h);
  }
  final_ln<<<1, 256, 0, stream>>>(h, fc_g, out);
}
